// GNS_68762426409211
// MI455X (gfx1250) — compile-verified, matrix operand order checked by reading, not run on hardware
//
#include <hip/hip_runtime.h>
#include <math.h>

#define NBUS  131072
#define NLINE 524288
#define NGEN  32768
#define KIT   30

#define SINW 20   // input tile row stride (dwords): 16B aligned, conflict-free
#define SHW  36   // hidden tile row stride (dwords): 16B aligned, conflict-free

// ---------------------------------------------------------------- WMMA types
typedef __attribute__((ext_vector_type(16))) __bf16 v16bf;
typedef __attribute__((ext_vector_type(8)))  float  v8f;
typedef __attribute__((ext_vector_type(4)))  float  vf4;
typedef __attribute__((ext_vector_type(4)))  unsigned vu4;

union FragA { unsigned u[8]; v16bf v; };

// ------------------------------------------------- global (AS1) load helpers
typedef const float __attribute__((address_space(1)))* gfp;
typedef const vf4   __attribute__((address_space(1)))* gf4p;
typedef const vu4   __attribute__((address_space(1)))* gu4p;
typedef const int   __attribute__((address_space(1)))* gip;

__device__ __forceinline__ float gldf (const float* p){ return *(gfp)(unsigned long long)p; }
__device__ __forceinline__ vf4   gldf4(const float* p){ return *(gf4p)(unsigned long long)p; }
__device__ __forceinline__ vu4   gldu4(const unsigned* p){ return *(gu4p)(unsigned long long)p; }
__device__ __forceinline__ int   gldi (const int* p){ return *(gip)(unsigned long long)p; }

// ------------------------------------------------------------ bf16 packing
__device__ __forceinline__ unsigned short f2bf(float f){
  unsigned u = __float_as_uint(f);
  u += 0x7FFFu + ((u >> 16) & 1u);
  return (unsigned short)(u >> 16);
}
#if __has_builtin(__builtin_amdgcn_cvt_pk_bf16_f32)
typedef __attribute__((ext_vector_type(2))) __bf16 v2bf;
__device__ __forceinline__ unsigned pack_bf16(float lo, float hi){
  union { v2bf v; unsigned u; } cv;
  cv.v = __builtin_amdgcn_cvt_pk_bf16_f32(lo, hi);
  return cv.u;
}
#else
__device__ __forceinline__ unsigned pack_bf16(float lo, float hi){
  return (unsigned)f2bf(lo) | ((unsigned)f2bf(hi) << 16);
}
#endif
__device__ __forceinline__ unsigned short bf1(float x){
  return (unsigned short)(pack_bf16(x, 0.f) & 0xFFFFu);
}
__device__ __forceinline__ float bf2f(unsigned short h){
  return __uint_as_float(((unsigned)h) << 16);
}
__device__ __forceinline__ float leaky(float x){ return x >= 0.f ? x : 0.01f*x; }

// --------------------------------------------------- wave-scope LDS ordering
__device__ __forceinline__ void wsync(){
#if __has_builtin(__builtin_amdgcn_wave_barrier) && __has_builtin(__builtin_amdgcn_fence)
  __builtin_amdgcn_fence(__ATOMIC_RELEASE, "wavefront");
  __builtin_amdgcn_wave_barrier();
  __builtin_amdgcn_fence(__ATOMIC_ACQUIRE, "wavefront");
#else
  __syncthreads();
#endif
}

// ---------------------------------------------------------------- WMMA core
__device__ __forceinline__ v8f wmma_bf16(FragA a, FragA b, v8f c){
  return __builtin_amdgcn_wmma_f32_16x16x32_bf16(false, a.v, false, b.v,
                                                 (short)0, c, false, false);
}
// B fragment: 2 x global_load_b128 per lane from packed weights
__device__ __forceinline__ void load_bfrag(FragA& b, const unsigned* f, int lane){
  vu4 x0 = gldu4(f + lane*8);
  vu4 x1 = gldu4(f + lane*8 + 4);
  b.u[0]=x0.x; b.u[1]=x0.y; b.u[2]=x0.z; b.u[3]=x0.w;
  b.u[4]=x1.x; b.u[5]=x1.y; b.u[6]=x1.z; b.u[7]=x1.w;
}
// A fragment from an LDS row (stride-aligned): 2 x ds_load_b128 per lane.
// half0 holds K pairs 0..7 (dw 0..3) and 16..23 (dw 8..11); half1 the others.
__device__ __forceinline__ void load_afrag(FragA& a, const unsigned* rowp, int half){
  const vu4* p = (const vu4*)rowp;
  vu4 x0 = p[half];
  vu4 x1 = p[2 + half];
  a.u[0]=x0.x; a.u[1]=x0.y; a.u[2]=x0.z; a.u[3]=x0.w;
  a.u[4]=x1.x; a.u[5]=x1.y; a.u[6]=x1.z; a.u[7]=x1.w;
}

// layer1 (pad-32 -> 64, leaky) + layer2 (64 -> 64, leaky); H2 left in sH.
// wp frags: [0..3]=W1 ntiles, [4..11]=W2 (ktile*4+ntile), [12..13]=W3 (64x16)
__device__ __forceinline__ void mlp_hidden(FragA aIn, const unsigned* wp,
                                           const float* b1, const float* b2,
                                           unsigned* sH, int lane)
{
  int row = lane & 15, half = lane >> 4;
  wsync();
  for (int nt = 0; nt < 4; ++nt){
    FragA b; load_bfrag(b, wp + nt*256, lane);
    v8f c = {0.f,0.f,0.f,0.f,0.f,0.f,0.f,0.f};
    c = wmma_bf16(aIn, b, c);
    float bias = gldf(b1 + nt*16 + row);
    for (int j = 0; j < 8; ++j){
      int M = j + 8*half;
      ((unsigned short*)(sH + M*SHW))[nt*16 + row] = bf1(leaky(c[j] + bias));
    }
  }
  wsync();
  FragA a2[2];
  load_afrag(a2[0], sH + row*SHW,      half);
  load_afrag(a2[1], sH + row*SHW + 16, half);
  wsync();
  for (int nt = 0; nt < 4; ++nt){
    v8f c = {0.f,0.f,0.f,0.f,0.f,0.f,0.f,0.f};
    for (int ch = 0; ch < 2; ++ch){
      FragA b; load_bfrag(b, wp + (4 + ch*4 + nt)*256, lane);
      c = wmma_bf16(a2[ch], b, c);
    }
    float bias = gldf(b2 + nt*16 + row);
    for (int j = 0; j < 8; ++j){
      int M = j + 8*half;
      ((unsigned short*)(sH + M*SHW))[nt*16 + row] = bf1(leaky(c[j] + bias));
    }
  }
  wsync();
}

// layer3 dout=1: f32 dot over 64 hidden, halves combined via shfl
__device__ __forceinline__ float mlp_out1(const unsigned* sH, const float* w3,
                                          const float* b3, int lane){
  int row = lane & 15, half = lane >> 4;
  const unsigned short* hr = (const unsigned short*)(sH + row*SHW);
  float s = 0.f;
  for (int j = 0; j < 32; ++j) s += bf2f(hr[32*half + j]) * gldf(w3 + 32*half + j);
  s += __shfl_xor(s, 16, 32);
  return s + gldf(b3);
}
// layer3 dout=16 (L_m): two WMMAs, D = H2(16x64) x W3(64x16)
__device__ __forceinline__ v8f mlp_out16(const unsigned* sH, const unsigned* wp,
                                         int lane){
  int row = lane & 15, half = lane >> 4;
  FragA a3[2];
  load_afrag(a3[0], sH + row*SHW,      half);
  load_afrag(a3[1], sH + row*SHW + 16, half);
  v8f c = {0.f,0.f,0.f,0.f,0.f,0.f,0.f,0.f};
  for (int ch = 0; ch < 2; ++ch){
    FragA b; load_bfrag(b, wp + (12 + ch)*256, lane);
    c = wmma_bf16(a3[ch], b, c);
  }
  return c;
}
__device__ __forceinline__ int kpair(int half, int vv){
  return (vv < 4) ? (8*half + 2*vv) : (16 + 8*half + 2*(vv-4));
}

// ---------------------------------------------------------------- small utils
__global__ void k_zero(float* p, int n){
  int i = blockIdx.x*256 + threadIdx.x;
  if (i < n) p[i] = 0.f;
}
__global__ void k_decode(const float* lines, int* isrc, int* idst){
  int e = blockIdx.x*256 + threadIdx.x;
  if (e < NLINE){
    isrc[e] = (int)gldf(lines + (size_t)e*7 + 0) - 1;
    idst[e] = (int)gldf(lines + (size_t)e*7 + 1) - 1;
  }
}
__global__ void k_init_bus(float* v, float* theta, int* nongen){
  int i = blockIdx.x*256 + threadIdx.x;
  if (i < NBUS){ v[i] = 1.f; theta[i] = 0.f; nongen[i] = 1; }
}
__global__ void k_gen_init(const float* gens, float* v, int* nongen,
                           float* dp, float* dq){
  int g = blockIdx.x*256 + threadIdx.x;
  if (g < NGEN){
    const float* gr = gens + (size_t)g*7;
    int b = (int)gldf(gr) - 1;
    v[b] = gldf(gr + 3);
    nongen[b] = 0;
    atomicAdd(&dp[b], gldf(gr + 1));
    atomicAdd(&dq[b], gldf(gr + 2));
  }
}
__global__ void k_init_dpdq(const float* buses, const float* v, float* dp, float* dq){
  int i = blockIdx.x*256 + threadIdx.x;
  if (i < NBUS){
    vf4 br = gldf4(buses + (size_t)i*4);
    float vv = v[i]*v[i];
    dp[i] = dp[i] - br.x - br.z*vv;
    dq[i] = dq[i] - br.y + br.w*vv;
  }
}
__global__ void k_gen_sums(const float* gens, float* scal){
  __shared__ float r0[256], r1[256], r2[256];
  int g = blockIdx.x*256 + threadIdx.x;
  float a=0.f, b=0.f, c=0.f;
  if (g < NGEN){
    const float* gr = gens + (size_t)g*7;
    a = gldf(gr + 4); b = gldf(gr + 5); c = gldf(gr + 6);
  }
  r0[threadIdx.x]=a; r1[threadIdx.x]=b; r2[threadIdx.x]=c;
  __syncthreads();
  for (int off = 128; off; off >>= 1){
    if (threadIdx.x < off){
      r0[threadIdx.x]+=r0[threadIdx.x+off];
      r1[threadIdx.x]+=r1[threadIdx.x+off];
      r2[threadIdx.x]+=r2[threadIdx.x+off];
    }
    __syncthreads();
  }
  if (threadIdx.x==0){
    atomicAdd(&scal[4], r0[0]); atomicAdd(&scal[5], r1[0]); atomicAdd(&scal[6], r2[0]);
  }
}

// ---------------------------------------------------------------- weight pack
struct PackArgs {
  const float* W1[6]; const float* W2[6]; const float* W3[6];
  unsigned* out;
};
__global__ void k_pack(PackArgs P){
  int idx = blockIdx.x*256 + threadIdx.x;
  const int total = KIT*6*3584;
  if (idx >= total) return;
  int it  = idx / (6*3584);
  int r   = idx % (6*3584);
  int mlp = r / 3584;
  int fo  = r % 3584;
  int frag = fo >> 8, within = fo & 255;
  int lane = within >> 3, vv = within & 7;
  int half = lane >> 4, nl = lane & 15;
  int kb = kpair(half, vv);
  float lo = 0.f, hi = 0.f;
  if (frag < 4){                                    // W1: 21(pad32) x 64
    int n = frag*16 + nl; int k = kb;
    const float* W = P.W1[mlp] + (size_t)it*21*64;
    if (k   < 21) lo = gldf(W + k*64 + n);
    if (k+1 < 21) hi = gldf(W + (k+1)*64 + n);
  } else if (frag < 12){                            // W2: 64 x 64
    int f = frag - 4; int ch = f >> 2, nt = f & 3;
    int n = nt*16 + nl; int k = ch*32 + kb;
    const float* W = P.W2[mlp] + (size_t)it*4096;
    lo = gldf(W + k*64 + n); hi = gldf(W + (k+1)*64 + n);
  } else {                                          // W3: 64 x dout (pad 16)
    int ch = frag - 12; int k = ch*32 + kb;
    int dout = (mlp == 5) ? 16 : 1;
    const float* W = P.W3[mlp] + (size_t)it*64*dout;
    if (nl < dout){ lo = gldf(W + k*dout + nl); hi = gldf(W + (k+1)*dout + nl); }
  }
  P.out[idx] = pack_bf16(lo, hi);
}

// ---------------------------------------------------------------- edge MLPs
struct EdgeArgs {
  const float* lines; const int* idst; const float* m;
  const unsigned* wp_iter;
  const float *b1[3], *b2[3], *w3[3], *b3[3];
  float* sacc[3];                                   // sv, st, sm
};
__global__ __launch_bounds__(256) void k_edge_mlp(EdgeArgs A){
  __shared__ unsigned sIn[8][16][SINW];
  __shared__ unsigned sHH[8][16][SHW];
  int wave = threadIdx.x >> 5, lane = threadIdx.x & 31;
  int row = lane & 15, half = lane >> 4;
  int e = (blockIdx.x*8 + wave)*16 + row;
  int d = gldi(A.idst + e);
  unsigned* tin = &sIn[wave][0][0];
  unsigned* th  = &sHH[wave][0][0];
  if (half == 0){                          // cols 0..15 : m[dst] (4x b128)
    const float* mr = A.m + (size_t)d*16;
    for (int j = 0; j < 4; ++j){
      vf4 q = gldf4(mr + 4*j);
      tin[row*SINW + 2*j]     = pack_bf16(q.x, q.y);
      tin[row*SINW + 2*j + 1] = pack_bf16(q.z, q.w);
    }
  } else {                                 // cols 16..20 : edge_feat, 21..31 : 0
    const float* lr = A.lines + (size_t)e*7;
    tin[row*SINW + 8]  = pack_bf16(gldf(lr + 2), gldf(lr + 3));
    tin[row*SINW + 9]  = pack_bf16(gldf(lr + 4), gldf(lr + 5));
    tin[row*SINW + 10] = pack_bf16(gldf(lr + 6), 0.f);
    for (int j = 11; j < 16; ++j) tin[row*SINW + j] = 0u;
  }
  wsync();
  FragA a;
  load_afrag(a, tin + row*SINW, half);
  for (int mlp = 0; mlp < 3; ++mlp){
    const unsigned* wp = A.wp_iter + mlp*3584;
    if (mlp < 2) __builtin_prefetch(A.wp_iter + (mlp+1)*3584, 0, 3);
    mlp_hidden(a, wp, A.b1[mlp], A.b2[mlp], th, lane);
    float out = mlp_out1(th, A.w3[mlp], A.b3[mlp], lane);
    if (half == 0) atomicAdd(&A.sacc[mlp][d], out);
  }
}

// ---------------------------------------------------------------- node MLPs
struct NodeArgs {
  float* v; float* theta; float* m;
  const float* dp; const float* dq;
  const float* sin_th; const float* sin_v; const float* sin_m;
  const int* nongen;
  const unsigned* wp_iter;                 // mlps 3,4,5 (L_theta, L_v, L_m)
  const float *b1[3], *b2[3];
  const float *w3t, *b3t, *w3v, *b3v, *b3m;
};
__global__ __launch_bounds__(256) void k_node_mlp(NodeArgs A){
  __shared__ unsigned sIn[8][16][SINW];
  __shared__ unsigned sHH[8][16][SHW];
  int wave = threadIdx.x >> 5, lane = threadIdx.x & 31;
  int row = lane & 15, half = lane >> 4;
  int bi = (blockIdx.x*8 + wave)*16 + row;
  unsigned* tin = &sIn[wave][0][0];
  unsigned* th  = &sHH[wave][0][0];
  float vb = gldf(A.v + bi), tb = gldf(A.theta + bi);
  const float* mr = A.m + (size_t)bi*16;
  if (half == 0){                          // cols 0..15 : v,theta,dp,dq,m0..11
    tin[row*SINW + 0] = pack_bf16(vb, tb);
    tin[row*SINW + 1] = pack_bf16(gldf(A.dp + bi), gldf(A.dq + bi));
    for (int j = 0; j < 3; ++j){
      vf4 q = gldf4(mr + 4*j);
      tin[row*SINW + 2 + 2*j]   = pack_bf16(q.x, q.y);
      tin[row*SINW + 3 + 2*j]   = pack_bf16(q.z, q.w);
    }
  } else {                                 // cols 16..19 : m12..15, 20 : s
    vf4 q = gldf4(mr + 12);
    tin[row*SINW + 8] = pack_bf16(q.x, q.y);
    tin[row*SINW + 9] = pack_bf16(q.z, q.w);
    for (int j = 10; j < 16; ++j) tin[row*SINW + j] = 0u;
  }
  wsync();
  const float* sins[3] = { A.sin_th, A.sin_v, A.sin_m };
  for (int mlp = 0; mlp < 3; ++mlp){
    if (half == 1) tin[row*SINW + 10] = pack_bf16(gldf(sins[mlp] + bi), 0.f);
    wsync();
    FragA a;
    load_afrag(a, tin + row*SINW, half);
    const unsigned* wp = A.wp_iter + (3 + mlp)*3584;
    if (mlp < 2) __builtin_prefetch(A.wp_iter + (4 + mlp)*3584, 0, 3);
    mlp_hidden(a, wp, A.b1[mlp], A.b2[mlp], th, lane);
    if (mlp == 0){                                   // L_theta
      float out = mlp_out1(th, A.w3t, A.b3t, lane);
      if (half == 0) A.theta[bi] = tb + out;
    } else if (mlp == 1){                            // L_v (non-gen gated)
      float out = mlp_out1(th, A.w3v, A.b3v, lane);
      if (half == 0 && gldi(A.nongen + bi)) A.v[bi] = vb + out;
    } else {                                         // L_m (WMMA 64->16)
      v8f c = mlp_out16(th, wp, lane);
      int n = row; float bias = gldf(A.b3m + n);
      int bus0 = (blockIdx.x*8 + wave)*16;
      for (int j = 0; j < 8; ++j){
        int M = j + 8*half;
        size_t idx = (size_t)(bus0 + M)*16 + n;
        A.m[idx] = gldf(A.m + idx) + c[j] + bias;
      }
    }
    wsync();
  }
}

// ---------------------------------------------------------------- edge trig
__global__ void k_edge_trig(const float* lines, const int* isrc, const int* idst,
                            const float* v, const float* theta,
                            float* pf, float* pt, float* qf, float* qt, float* scal){
  int e = blockIdx.x*256 + threadIdx.x;
  float msg = 0.f;
  if (e < NLINE){
    int s = gldi(isrc + e), d = gldi(idst + e);
    // faithful to reference quirk: y/tau/shift/b/dij indexed at bus indices
    const float* ls = lines + (size_t)s*7;
    const float* ld = lines + (size_t)d*7;
    float lr2 = gldf(ls+2), lr3 = gldf(ls+3);
    float ys   = rsqrtf(lr2*lr2 + lr3*lr3);
    float bs = gldf(ls+4), taus = gldf(ls+5), shs = gldf(ls+6);
    float dr2 = gldf(ld+2), dr3 = gldf(ld+3);
    float yd   = rsqrtf(dr2*dr2 + dr3*dr3);
    float bd = gldf(ld+4), taud = gldf(ld+5), shd = gldf(ld+6);
    float dijs = gldf(theta + gldi(isrc + s)) - gldf(theta + gldi(idst + s));
    float djid = gldf(theta + gldi(idst + d)) - gldf(theta + gldi(isrc + d));
    float vs = gldf(v + s), vd = gldf(v + d);
    float ts = gldf(theta + s), td = gldf(theta + d);
    float a1 = ts - td - dijs - shs;
    float a2 = td - ts - dijs + shs;
    float a3 = td - ts - djid - shd;
    float sdij = __sinf(dijs), cdij = __cosf(dijs), sdji = __sinf(djid);
    float g1 = vs*vd*ys/taus, g2 = vd*vs*yd/taud;
    float it = vs/taus;
    msg = fabsf(g1*(__sinf(a1) + __sinf(a2))
                + vs/(taus*taus)*ys*sdij + vd*vd*ys*sdij);
    float p_from = g1*__sinf(a1) + it*it*ys*sdij;
    float p_to   = g2*__sinf(a3) + vd*vd*yd*sdji;
    float q_from = -g1*__cosf(a1) + it*it*(ys*cdij - 0.5f*bs);
    float q_to   = -g2*__cosf(a3) + vd*vd*(yd*sdji - 0.5f*bd);
    atomicAdd(&pf[d], p_from);
    atomicAdd(&pt[s], p_to);
    atomicAdd(&qf[d], q_from);
    atomicAdd(&qt[s], q_to);
  }
  __shared__ float red[256];
  red[threadIdx.x] = msg;
  __syncthreads();
  for (int off = 128; off; off >>= 1){
    if (threadIdx.x < off) red[threadIdx.x] += red[threadIdx.x + off];
    __syncthreads();
  }
  if (threadIdx.x == 0) atomicAdd(&scal[0], red[0]);   // p_joule
}

__global__ void k_bus_term(const float* buses, const float* v, float* scal){
  int i = blockIdx.x*256 + threadIdx.x;
  float t = 0.f;
  if (i < NBUS){
    vf4 br = gldf4(buses + (size_t)i*4);
    float vi = gldf(v + i);
    t = br.x + vi*vi*br.z;
  }
  __shared__ float red[256];
  red[threadIdx.x] = t;
  __syncthreads();
  for (int off = 128; off; off >>= 1){
    if (threadIdx.x < off) red[threadIdx.x] += red[threadIdx.x + off];
    __syncthreads();
  }
  if (threadIdx.x == 0) atomicAdd(&scal[1], red[0]);
}

__global__ void k_lam(float* scal){
  float pg = scal[0] + scal[1];
  float pgset = scal[4], pmin = scal[5], pmax = scal[6];
  scal[2] = (pg < pgset) ? (pg - pmin) / (2.f*(pgset - pmin))
                         : (pg - 2.f*pgset + pmax) / (2.f*(pmax - pgset));
}

__global__ void k_gen_pg(const float* gens, const float* scal, float* genP){
  int g = blockIdx.x*256 + threadIdx.x;
  if (g < NGEN){
    const float* gr = gens + (size_t)g*7;
    float lam = gldf(scal + 2);
    float pgset = gldf(gr+4), pmin = gldf(gr+5), pmax = gldf(gr+6);
    float pg = (lam < 0.5f) ? pmin + 2.f*(pgset - pmin)*lam
                            : 2.f*pgset - pmax + 2.f*(pmax - pgset)*lam;
    atomicAdd(&genP[(int)gldf(gr) - 1], pg);
  }
}

__global__ void k_imbalance(const float* buses, const float* v,
                            const float* pf, const float* pt,
                            const float* qf, const float* qt,
                            const float* genP, float* dp, float* dq,
                            float* scal, float wk){
  int i = blockIdx.x*256 + threadIdx.x;
  float l = 0.f;
  if (i < NBUS){
    vf4 br = gldf4(buses + (size_t)i*4);
    float vi = gldf(v + i);
    float vv = vi*vi;
    float qfv = gldf(qf + i), qtv = gldf(qt + i);
    float qg_new = br.y - br.w*vv - qfv - qtv;
    float dpn = gldf(genP + i) - br.x - br.z*vv + gldf(pf + i) + gldf(pt + i);
    float dqn = (qg_new - br.y + br.w*vv) + qfv + qtv;
    dp[i] = dpn; dq[i] = dqn;
    l = wk*(dpn*dpn + dqn*dqn);
  }
  __shared__ float red[256];
  red[threadIdx.x] = l;
  __syncthreads();
  for (int off = 128; off; off >>= 1){
    if (threadIdx.x < off) red[threadIdx.x] += red[threadIdx.x + off];
    __syncthreads();
  }
  if (threadIdx.x == 0) atomicAdd(&scal[3], red[0]);
}

__global__ void k_final(const float* v, const float* theta, const float* scal,
                        float* out){
  int i = blockIdx.x*256 + threadIdx.x;
  if (i < NBUS){ out[i] = gldf(v + i); out[NBUS + i] = gldf(theta + i); }
  if (blockIdx.x == 0 && threadIdx.x == 0) out[2*NBUS] = gldf(scal + 3);
}

// ---------------------------------------------------------------- host driver
extern "C" void kernel_launch(void* const* d_in, const int* in_sizes, int n_in,
                              void* d_out, int out_size, void* d_ws, size_t ws_size,
                              hipStream_t stream){
  (void)in_sizes; (void)n_in; (void)out_size; (void)ws_size;
  const float* buses = (const float*)d_in[0];
  const float* lines = (const float*)d_in[1];
  const float* gens  = (const float*)d_in[2];
  auto prm = [&](int mlp, int which)->const float* {
    return (const float*)d_in[3 + mlp*6 + which];
  };

  char* base = (char*)d_ws;
  size_t off = 0;
  auto alloc = [&](size_t bytes)->char* {
    off = (off + 255) & ~(size_t)255;
    char* p = base + off; off += bytes; return p;
  };
  float* scal   = (float*)alloc(64*sizeof(float));
  int*   isrc   = (int*)alloc((size_t)NLINE*sizeof(int));
  int*   idst   = (int*)alloc((size_t)NLINE*sizeof(int));
  float* v      = (float*)alloc((size_t)NBUS*sizeof(float));
  float* theta  = (float*)alloc((size_t)NBUS*sizeof(float));
  float* dp     = (float*)alloc((size_t)NBUS*sizeof(float));
  float* dq     = (float*)alloc((size_t)NBUS*sizeof(float));
  float* m      = (float*)alloc((size_t)NBUS*16*sizeof(float));
  float* sblk   = (float*)alloc((size_t)3*NBUS*sizeof(float));
  float *sv = sblk, *st = sblk + NBUS, *sm = sblk + 2*NBUS;
  float* ablk   = (float*)alloc((size_t)5*NBUS*sizeof(float));
  float *pf = ablk, *pt = ablk + NBUS, *qf = ablk + 2*NBUS,
        *qt = ablk + 3*NBUS, *genP = ablk + 4*NBUS;
  int*   nongen = (int*)alloc((size_t)NBUS*sizeof(int));
  unsigned* wpack = (unsigned*)alloc((size_t)KIT*6*3584*sizeof(unsigned));

  // ---- one-time setup (re-done each call for determinism) ----
  k_zero<<<1, 256, 0, stream>>>(scal, 64);
  k_zero<<<(NBUS*16)/256, 256, 0, stream>>>(m, NBUS*16);
  k_zero<<<NBUS/256, 256, 0, stream>>>(dp, NBUS);
  k_zero<<<NBUS/256, 256, 0, stream>>>(dq, NBUS);
  k_decode<<<NLINE/256, 256, 0, stream>>>(lines, isrc, idst);

  PackArgs PA;
  for (int j = 0; j < 6; ++j){
    PA.W1[j] = prm(j,0); PA.W2[j] = prm(j,2); PA.W3[j] = prm(j,4);
  }
  PA.out = wpack;
  int ptot = KIT*6*3584;
  k_pack<<<(ptot + 255)/256, 256, 0, stream>>>(PA);

  k_init_bus<<<NBUS/256, 256, 0, stream>>>(v, theta, nongen);
  k_gen_init<<<NGEN/256, 256, 0, stream>>>(gens, v, nongen, dp, dq);
  k_init_dpdq<<<NBUS/256, 256, 0, stream>>>(buses, v, dp, dq);
  k_gen_sums<<<NGEN/256, 256, 0, stream>>>(gens, scal);

  // ---- K iterations ----
  for (int k = 0; k < KIT; ++k){
    k_zero<<<(3*NBUS)/256, 256, 0, stream>>>(sblk, 3*NBUS);
    k_zero<<<(5*NBUS)/256, 256, 0, stream>>>(ablk, 5*NBUS);
    k_zero<<<1, 256, 0, stream>>>(scal, 2);

    EdgeArgs EA;
    EA.lines = lines; EA.idst = idst; EA.m = m;
    EA.wp_iter = wpack + (size_t)k*6*3584;
    float* saccs[3] = { sv, st, sm };                 // phi_v, phi_theta, phi_m
    for (int j = 0; j < 3; ++j){
      EA.b1[j] = prm(j,1) + (size_t)k*64;
      EA.b2[j] = prm(j,3) + (size_t)k*64;
      EA.w3[j] = prm(j,4) + (size_t)k*64;
      EA.b3[j] = prm(j,5) + (size_t)k;
      EA.sacc[j] = saccs[j];
    }
    k_edge_mlp<<<NLINE/128, 256, 0, stream>>>(EA);

    NodeArgs NA;
    NA.v = v; NA.theta = theta; NA.m = m; NA.dp = dp; NA.dq = dq;
    NA.sin_th = st; NA.sin_v = sv; NA.sin_m = sm; NA.nongen = nongen;
    NA.wp_iter = wpack + (size_t)k*6*3584;
    for (int j = 0; j < 3; ++j){                      // L_theta, L_v, L_m
      NA.b1[j] = prm(3+j,1) + (size_t)k*64;
      NA.b2[j] = prm(3+j,3) + (size_t)k*64;
    }
    NA.w3t = prm(3,4) + (size_t)k*64; NA.b3t = prm(3,5) + (size_t)k;
    NA.w3v = prm(4,4) + (size_t)k*64; NA.b3v = prm(4,5) + (size_t)k;
    NA.b3m = prm(5,5) + (size_t)k*16;
    k_node_mlp<<<NBUS/128, 256, 0, stream>>>(NA);

    k_edge_trig<<<NLINE/256, 256, 0, stream>>>(lines, isrc, idst, v, theta,
                                               pf, pt, qf, qt, scal);
    k_bus_term<<<NBUS/256, 256, 0, stream>>>(buses, v, scal);
    k_lam<<<1, 1, 0, stream>>>(scal);
    k_gen_pg<<<NGEN/256, 256, 0, stream>>>(gens, scal, genP);
    float wk = powf(0.9f, (float)(KIT - k)) / (float)NBUS;
    k_imbalance<<<NBUS/256, 256, 0, stream>>>(buses, v, pf, pt, qf, qt,
                                              genP, dp, dq, scal, wk);
  }
  k_final<<<NBUS/256, 256, 0, stream>>>(v, theta, scal, (float*)d_out);
}
